// LeafBlockAttention_249108103788
// MI455X (gfx1250) — matrix operand order repaired and checked by reading, not compiled, not run on hardware
//
#include <hip/hip_runtime.h>
#include <hip/hip_bf16.h>
#include <stdint.h>

#define DIMC 256
#define LQ   16
#define NH   8
#define HD   32
#define KCNT 17
#define BATCH 2
#define NSEQ 32768
#define NBLK 2048
#define SCALE 0.17677669529663689f   // 32^-0.5

// LDS row strides chosen so lanes 0..15 map to distinct banks on tile loads
#define XS  260   // pad for 256-wide f16 rows
#define VTS 36    // pad for 32-wide f16 rows (v^T, combined)

typedef __attribute__((ext_vector_type(16))) _Float16 v16h;
typedef __attribute__((ext_vector_type(8)))  float    v8f;

union V16 { v16h v; _Float16 h[16]; uint32_t u[8]; };

// ---------------------------------------------------------------------------
// Pre-pass: convert Wqkv [768x256] and Wproj [256x256] (f32, row-major W[n][k])
// into f16 WMMA-B tiles in d_ws, pre-swizzled so each lane reads its 8 dwords
// contiguously. Tile t = nt*8+kt covers N cols nt*16..+15, K rows kt*32..+31.
// Lane l holds column n = l%16; dword v holds K pair (kb+2v, kb+2v+1),
// kb = (l>=16 ? 16 : 0).  (contiguous K order, same as load_a_tile)
// ---------------------------------------------------------------------------
__global__ __launch_bounds__(256)
void convert_weights_k(const float* __restrict__ Wqkv,
                       const float* __restrict__ Wproj,
                       uint32_t* __restrict__ ws) {
  const int t   = blockIdx.x;       // 0..511 (384 qkv tiles + 128 proj tiles)
  const int tid = threadIdx.x;      // 0..255
  const int v    = tid & 7;
  const int lane = tid >> 3;
  const float* W;
  int nt, kt;
  if (t < 384) { W = Wqkv;  nt = t >> 3;        kt = t & 7; }
  else         { W = Wproj; nt = (t - 384) >> 3; kt = (t - 384) & 7; }
  const int n = nt * 16 + (lane & 15);
  const int k = kt * 32 + ((lane >= 16) ? 16 : 0) + 2 * v;
  union { _Float16 h[2]; uint32_t u; } pk;
  pk.h[0] = (_Float16)W[n * DIMC + k];
  pk.h[1] = (_Float16)W[n * DIMC + k + 1];
  ws[(size_t)t * 256 + tid] = pk.u;
}

// Load a 16x32 f16 A tile from an LDS row-major array (row stride = stride,
// per-lane row m = lane%16).  Contiguous K: v0..7 -> K {2*hi8+2v, +1},
// hi8 = (lane>=16)?8:0.
__device__ __forceinline__ v16h load_a_tile(const _Float16* row, int hi8) {
  V16 A;
#pragma unroll
  for (int vv = 0; vv < 8; ++vv) {
    const int k = (hi8 << 1) + (vv << 1);
    A.h[2 * vv]     = row[k];
    A.h[2 * vv + 1] = row[k + 1];
  }
  return A.v;
}

// Load a 32x16 f16 B tile from LDS where the lane's column is contiguous in K
// (pointer col -> element K=0).  kb = (lane>=16)?16:0.
__device__ __forceinline__ v16h load_b_tile(const _Float16* col, int kb) {
  V16 Bv;
#pragma unroll
  for (int vv = 0; vv < 8; ++vv) {
    const int k = kb + 2 * vv;
    Bv.h[2 * vv]     = col[k];
    Bv.h[2 * vv + 1] = col[k + 1];
  }
  return Bv.v;
}

// ---------------------------------------------------------------------------
// Main kernel: one workgroup (8 wave32) per leaf block.
// ---------------------------------------------------------------------------
__global__ __launch_bounds__(256)
void leafblk_attn_k(const float* __restrict__ x,
                    const int*   __restrict__ amask,
                    const float* __restrict__ ef,
                    const float* __restrict__ bqkv,
                    const float* __restrict__ bproj,
                    const float* __restrict__ Wgate,
                    const float* __restrict__ bgate,
                    const v16h*  __restrict__ wsQ,   // 384 preswizzled B tiles
                    const v16h*  __restrict__ wsP,   // 128 preswizzled B tiles
                    float* __restrict__ out) {
  __shared__ _Float16 x16[LQ][XS];        // A for QKV GEMM
  __shared__ _Float16 q16[LQ][XS];        // q (f16)
  __shared__ _Float16 k16[KCNT][XS];      // k rows 0..16 (f16)
  __shared__ _Float16 vt16[DIMC][VTS];    // v transposed [col][k 0..16, pad->0]
  __shared__ float    sc[NH][LQ][KCNT];   // raw qk scores
  __shared__ _Float16 comb[NH][LQ][VTS];  // softmax+gate, K padded to 32
  __shared__ _Float16 a16[LQ][XS];        // attention output (A for proj)

  const int tid  = threadIdx.x;
  const int lane = tid & 31;
  const int w    = tid >> 5;              // wave id == head id in attn phases
  const int l16  = lane & 15;
  const int hi8  = (lane >= 16) ? 8 : 0;  // A-tile K offset AND C-tile M offset
  const int kb   = (lane >= 16) ? 16 : 0; // B-tile K offset
  const int g    = blockIdx.x;            // 0..B*NB-1
  const int b    = g / NBLK;
  const int blk  = g % NBLK;

  // ---- Phase 0: stage x block (f32 -> f16), zero v^T pad -------------------
  {
    const float* xr = x + ((size_t)b * NSEQ + (size_t)blk * LQ) * DIMC;
#pragma unroll
    for (int i = 0; i < LQ; ++i)
      x16[i][tid] = (_Float16)xr[i * DIMC + tid];
#pragma unroll
    for (int kk = 0; kk < VTS; ++kk)
      vt16[tid][kk] = (_Float16)0.f;
  }
  __syncthreads();

  // ---- Phase 1: fused QKV GEMM  [16x256] x [256x768] -----------------------
  {
    v8f acc[6] = {};
    for (int kt = 0; kt < 8; ++kt) {
      const v16h A = load_a_tile(&x16[l16][kt * 32], hi8);
#pragma unroll
      for (int j = 0; j < 6; ++j) {
        const int tn = w + 8 * j;                 // 48 N-tiles over 8 waves
        const v16h Bv = wsQ[(tn * 8 + kt) * 32 + lane];
        acc[j] = __builtin_amdgcn_wmma_f32_16x16x32_f16(
            false, A, false, Bv, (short)0, acc[j], false, false);
      }
    }
#pragma unroll
    for (int j = 0; j < 6; ++j) {
      const int nc  = (w + 8 * j) * 16 + l16;     // 0..767
      const float bia = bqkv[nc];
#pragma unroll
      for (int r = 0; r < 8; ++r) {
        const int m = r + hi8;
        const float val = acc[j][r] + bia;
        if (nc < 256)      q16[m][nc]        = (_Float16)val;
        else if (nc < 512) k16[m][nc - 256]  = (_Float16)val;
        else               vt16[nc - 512][m] = (_Float16)val;  // v transposed
      }
    }
  }
  __syncthreads();

  // ---- Phase 1b: block-node row = mean of leaf rows (linearity) ------------
  {
    const int c = tid;                 // one column of k and one of v each
    float sk = 0.f, sv = 0.f;
#pragma unroll
    for (int m = 0; m < LQ; ++m) { sk += (float)k16[m][c]; sv += (float)vt16[c][m]; }
    k16[LQ][c]  = (_Float16)(sk * 0.0625f);
    vt16[c][LQ] = (_Float16)(sv * 0.0625f);
  }
  __syncthreads();

  // ---- Phase 2: per-head QK^T (leaf cols via WMMA, block col via VALU) -----
  {
    const int h = w;
    const v16h A  = load_a_tile(&q16[l16][h * HD], hi8);
    const v16h Bv = load_b_tile(&k16[l16][h * HD], kb);   // B[d][n] = k[n][d]
    v8f accs = {};
    accs = __builtin_amdgcn_wmma_f32_16x16x32_f16(
        false, A, false, Bv, (short)0, accs, false, false);
#pragma unroll
    for (int r = 0; r < 8; ++r)
      sc[h][r + hi8][l16] = accs[r];
    if (lane < 16) {                   // column 16: dot with block-node key
      float s = 0.f;
      const _Float16* qr = &q16[lane][h * HD];
      const _Float16* kr = &k16[LQ][h * HD];
#pragma unroll
      for (int d = 0; d < HD; ++d) s += (float)qr[d] * (float)kr[d];
      sc[h][lane][LQ] = s;
    }
  }
  __syncthreads();

  // ---- Phase 3: scale + bias + mask + softmax + edge gate (128 rows) -------
  if (tid < NH * LQ) {
    const int h = tid >> 4, m = tid & 15;
    const int*   mrow = amask + ((size_t)blk * LQ + m) * KCNT;
    const float* erow = ef + ((size_t)blk * LQ + m) * KCNT * 4;
    const float g0 = Wgate[h * 4 + 0], g1 = Wgate[h * 4 + 1];
    const float g2 = Wgate[h * 4 + 2], g3 = Wgate[h * 4 + 3];
    const float gb = bgate[h];
    float s[KCNT], gt[KCNT];
    int   vis[KCNT];
    float mx = -3.0e38f;
#pragma unroll
    for (int kk = 0; kk < KCNT; ++kk) {
      vis[kk] = mrow[kk];
      float b0, b1, b2, b3;
      if (kk == LQ || kk == m) { b0 = b1 = b2 = 0.f; b3 = 1.f; }   // unit bias
      else { b0 = erow[kk * 4 + 0]; b1 = erow[kk * 4 + 1];
             b2 = erow[kk * 4 + 2]; b3 = erow[kk * 4 + 3]; }
      gt[kk] = g0 * b0 + g1 * b1 + g2 * b2 + g3 * b3 + gb;
      s[kk]  = vis[kk] ? (sc[h][m][kk] * SCALE + b3) : -3.0e38f;
      mx = fmaxf(mx, s[kk]);
    }
    float denom = 0.f;
#pragma unroll
    for (int kk = 0; kk < KCNT; ++kk) {
      s[kk] = vis[kk] ? __expf(s[kk] - mx) : 0.f;
      denom += s[kk];
    }
    const float inv = 1.f / denom;
#pragma unroll
    for (int kk = 0; kk < KCNT; ++kk)
      comb[h][m][kk] = (_Float16)(vis[kk] ? (s[kk] * inv + gt[kk]) : 0.f);
#pragma unroll
    for (int kk = KCNT; kk < 32; ++kk)
      comb[h][m][kk] = (_Float16)0.f;   // K-pad for WMMA
  }
  __syncthreads();

  // ---- Phase 4: per-head (attn+gate) @ V, K padded 17->32 ------------------
  {
    const int h = w;
    const v16h A = load_a_tile(&comb[h][l16][0], hi8);
#pragma unroll
    for (int dt = 0; dt < 2; ++dt) {
      const int nc = h * HD + dt * 16 + l16;
      const v16h Bv = load_b_tile(&vt16[nc][0], kb);   // v^T: K contiguous
      v8f accp = {};
      accp = __builtin_amdgcn_wmma_f32_16x16x32_f16(
          false, A, false, Bv, (short)0, accp, false, false);
#pragma unroll
      for (int r = 0; r < 8; ++r)
        a16[r + hi8][nc] = (_Float16)accp[r];
    }
  }
  __syncthreads();

  // ---- Phase 5: output projection [16x256] x [256x256] + bias -> global ----
  {
    v8f acc[2] = {};
    for (int kt = 0; kt < 8; ++kt) {
      const v16h A = load_a_tile(&a16[l16][kt * 32], hi8);
#pragma unroll
      for (int j = 0; j < 2; ++j) {
        const int tn = w + 8 * j;                 // 16 N-tiles over 8 waves
        const v16h Bv = wsP[(tn * 8 + kt) * 32 + lane];
        acc[j] = __builtin_amdgcn_wmma_f32_16x16x32_f16(
            false, A, false, Bv, (short)0, acc[j], false, false);
      }
    }
#pragma unroll
    for (int j = 0; j < 2; ++j) {
      const int nc  = (w + 8 * j) * 16 + l16;
      const float bia = bproj[nc];
#pragma unroll
      for (int r = 0; r < 8; ++r) {
        const int m = r + hi8;
        out[((size_t)b * NSEQ + (size_t)blk * LQ + m) * DIMC + nc] = acc[j][r] + bia;
      }
    }
  }
}

extern "C" void kernel_launch(void* const* d_in, const int* in_sizes, int n_in,
                              void* d_out, int out_size, void* d_ws, size_t ws_size,
                              hipStream_t stream) {
  (void)in_sizes; (void)n_in; (void)out_size; (void)ws_size;
  const float* x     = (const float*)d_in[0];
  const int*   amask = (const int*)  d_in[1];
  const float* ef    = (const float*)d_in[2];
  const float* Wqkv  = (const float*)d_in[3];
  const float* bqkv  = (const float*)d_in[4];
  const float* Wproj = (const float*)d_in[5];
  const float* bproj = (const float*)d_in[6];
  const float* Wgate = (const float*)d_in[7];
  const float* bgate = (const float*)d_in[8];
  float* out = (float*)d_out;
  uint32_t* ws = (uint32_t*)d_ws;

  // ws layout (dwords): [0,98304) qkv B-tiles, [98304,131072) proj B-tiles
  convert_weights_k<<<512, 256, 0, stream>>>(Wqkv, Wproj, ws);
  const v16h* wsQ = (const v16h*)ws;
  const v16h* wsP = (const v16h*)(ws + 98304);
  leafblk_attn_k<<<BATCH * NBLK, 256, 0, stream>>>(
      x, amask, ef, bqkv, bproj, Wgate, bgate, wsQ, wsP, out);
}
